// GConvF_58961311039798
// MI455X (gfx1250) — matrix operand order repaired and checked by reading, not compiled, not run on hardware
//
#include <hip/hip_runtime.h>

typedef __attribute__((ext_vector_type(16))) __bf16 v16bf;
typedef __attribute__((ext_vector_type(8)))  float  v8f;

#define NB  16
#define NC  64
#define NH  96
#define NW  96
#define NO  64
#define CP  48   // padded cin stride in LDS (16B-aligned rows, few bank conflicts)

// split an f32 value into bf16 hi (RNE) + bf16 lo (RNE of residual)
__device__ __forceinline__ void bf16_split(float v, unsigned short& hi, unsigned short& lo) {
    union { float f; unsigned u; } a; a.f = v;
    unsigned r = a.u + 0x7fffu + ((a.u >> 16) & 1u);
    unsigned short h = (unsigned short)(r >> 16);
    union { unsigned u; float f; } b; b.u = ((unsigned)h) << 16;
    float res = v - b.f;
    union { float f; unsigned u; } c; c.f = res;
    unsigned r2 = c.u + 0x7fffu + ((c.u >> 16) & 1u);
    hi = h;
    lo = (unsigned short)(r2 >> 16);
}

struct Pair128 { uint4 a, b; };

// A-fragment: lane half kh needs K = {kh*8..kh*8+7, 16+kh*8..16+kh*8+7}
__device__ __forceinline__ v16bf lds_loadA(const unsigned short* p) {
    Pair128 s;
    s.a = *(const uint4*)(p);        // +0  elements (16B)
    s.b = *(const uint4*)(p + 16);   // +32 bytes
    return __builtin_bit_cast(v16bf, s);
}

// B-fragment: lane half kh needs K = {kh*8..kh*8+7, 16+kh*8..16+kh*8+7}
__device__ __forceinline__ v16bf lds_loadB(const unsigned short* p) {
    Pair128 s;
    s.a = *(const uint4*)(p);        // +0  elements
    s.b = *(const uint4*)(p + 16);   // +32 bytes
    return __builtin_bit_cast(v16bf, s);
}

__global__ __launch_bounds__(96)
void GConvF_58961311039798_kernel(const float* __restrict__ xg,
                                  const float* __restrict__ wg,
                                  float* __restrict__ outg)
{
    // x slab: 3 rows x 50 pixels (16*3 outputs + 2 halo) x 32 cin (padded CP)
    __shared__ unsigned short xs_hi[3 * 50 * CP];
    __shared__ unsigned short xs_lo[3 * 50 * CP];
    // weight slab: 9 taps x 16 outch x 32 cin (padded CP)
    __shared__ unsigned short ws_hi[9 * 16 * CP];
    __shared__ unsigned short ws_lo[9 * 16 * CP];

    const int bx   = blockIdx.x;
    const int half = bx & 1;          // which half-row of W (48 outputs each)
    const int og   = (bx >> 1) & 3;   // 16-output-channel group
    const int t2   = bx >> 3;
    const int hrow = t2 % NH;
    const int b    = t2 / NH;

    const int tid  = threadIdx.x;
    const int wave = tid >> 5;        // 0..2 -> 16-pixel N-tile
    const int lane = tid & 31;
    const int n    = lane & 15;
    const int kh   = lane >> 4;

    v8f acc[4] = {v8f{}, v8f{}, v8f{}, v8f{}};   // one accumulator per rotation

    for (int ph = 0; ph < 2; ++ph) {
        if (ph) __syncthreads();      // previous phase compute done before restage
        const int cb = ph * 32;

        // ---- stage x (coalesced over w), convert f32 -> bf16 hi/lo ----
        for (int idx = tid; idx < 3 * 50 * 32; idx += 96) {
            const int wp = idx % 50;
            const int rw = (idx / 50) % 3;
            const int c  = idx / 150;
            const int w  = half * 48 + wp - 1;
            const int r  = hrow - 1 + rw;
            float v = 0.0f;
            if ((unsigned)w < (unsigned)NW && (unsigned)r < (unsigned)NH) {
                const size_t gi = (((size_t)b * NC + cb + c) * NH + r) * NW + w;
                v = xg[gi];
                if (ph == 0) {
                    // warm L2/GL1 for the next cin phase (global_prefetch_b8)
                    __builtin_prefetch(&xg[(((size_t)b * NC + 32 + c) * NH + r) * NW + w], 0, 1);
                }
            }
            unsigned short h16, l16;
            bf16_split(v, h16, l16);
            xs_hi[(rw * 50 + wp) * CP + c] = h16;
            xs_lo[(rw * 50 + wp) * CP + c] = l16;
        }

        // ---- stage weights for this output-channel group ----
        for (int idx = tid; idx < 9 * 16 * 32; idx += 96) {
            const int t = idx % 9;
            const int i = (idx / 9) % 32;
            const int o = idx / 288;
            const float v = wg[(((size_t)og * 16 + o) * NC + cb + i) * 9 + t];
            unsigned short h16, l16;
            bf16_split(v, h16, l16);
            ws_hi[(t * 16 + o) * CP + i] = h16;
            ws_lo[(t * 16 + o) * CP + i] = l16;
        }
        __syncthreads();

        // ---- implicit GEMM: 9 spatial taps x 4 rotations, bf16x3 split ----
        // ring (clockwise) then center; rotation r uses weight tap (s - 2r) mod 8
        constexpr int KY[9] = {0, 0, 0, 1, 2, 2, 2, 1, 1};
        constexpr int KX[9] = {0, 1, 2, 2, 2, 1, 0, 0, 1};
#pragma unroll
        for (int s = 0; s < 9; ++s) {
            const int wp  = wave * 16 + n + KX[s];
            const int xb  = (KY[s] * 50 + wp) * CP + kh * 8;
            const v16bf Bhi = lds_loadB(&xs_hi[xb]);
            const v16bf Blo = lds_loadB(&xs_lo[xb]);
#pragma unroll
            for (int ri = 0; ri < 4; ++ri) {
                const int wt = (s < 8) ? ((s - 2 * ri) & 7) : 8;
                const int ab = (wt * 16 + n) * CP + kh * 8;
                const v16bf Ahi = lds_loadA(&ws_hi[ab]);
                const v16bf Alo = lds_loadA(&ws_lo[ab]);
                acc[ri] = __builtin_amdgcn_wmma_f32_16x16x32_bf16(
                    false, Ahi, false, Bhi, (short)0, acc[ri], false, false);
                acc[ri] = __builtin_amdgcn_wmma_f32_16x16x32_bf16(
                    false, Ahi, false, Blo, (short)0, acc[ri], false, false);
                acc[ri] = __builtin_amdgcn_wmma_f32_16x16x32_bf16(
                    false, Alo, false, Bhi, (short)0, acc[ri], false, false);
            }
        }
    }

    // ---- store: 4 rotations are contiguous -> float4 (global_store_b128) ----
    const int w = half * 48 + wave * 16 + n;
#pragma unroll
    for (int j = 0; j < 8; ++j) {
        const int o = og * 16 + j + 8 * kh;   // C/D layout: VGPR j -> M=j (+8 for hi half)
        const size_t off = ((((size_t)b * NO + o) * NH + hrow) * NW + w) * 4;
        float4 v4 = make_float4(acc[0][j], acc[1][j], acc[2][j], acc[3][j]);
        *(float4*)(outg + off) = v4;
    }
}

extern "C" void kernel_launch(void* const* d_in, const int* in_sizes, int n_in,
                              void* d_out, int out_size, void* d_ws, size_t ws_size,
                              hipStream_t stream) {
    (void)in_sizes; (void)n_in; (void)d_ws; (void)ws_size; (void)out_size;
    const float* x = (const float*)d_in[0];
    const float* w = (const float*)d_in[1];
    float* out = (float*)d_out;
    dim3 grid(NB * NH * 4 * 2);   // (batch, row, outch-group, half-row) = 12288 blocks
    dim3 block(96);               // 3 wave32s, one 16-pixel N-tile each
    GConvF_58961311039798_kernel<<<grid, block, 0, stream>>>(x, w, out);
}
